// NMTDecoder_15401752723696
// MI455X (gfx1250) — matrix operand order repaired and checked by reading, not compiled, not run on hardware
//
#include <hip/hip_runtime.h>

// ---------------------------------------------------------------------------
// NMT decoder: GRU recurrence (64 sequential steps) + batched WMMA classifier
// Target: gfx1250 (CDNA5), wave32, v_wmma_f32_16x16x32_bf16 + TDM async tiles
// ---------------------------------------------------------------------------

typedef __bf16 bf16_t;
typedef __attribute__((ext_vector_type(16))) __bf16 v16bf;
typedef __attribute__((ext_vector_type(8)))  __bf16 v8bf;
typedef __attribute__((ext_vector_type(8)))  float  v8f;
typedef __attribute__((ext_vector_type(4)))  int    si4;
typedef __attribute__((ext_vector_type(8)))  int    si8;

#define DV 32000
#define DE 512
#define DH 1024
#define DB 32
#define DS 128
#define DT 64
#define D3H 3072
#define KIH 1536   // E + H
#define KCLS 2048  // 2H

// ---- fragment loaders (LDS row-major tiles, bf16) -------------------------
// A: 16x32 (MxK). lane<16: M=lane, K = k0 + {0..7, 16..23}; lane>=16: +8.
__device__ __forceinline__ v16bf frag_a(const bf16_t* buf, int ldk, int m_base,
                                        int k0, int lane) {
  const int half = (lane >> 4) & 1;
  const int m = m_base + (lane & 15);
  const bf16_t* p = buf + m * ldk + k0 + half * 8;
  v8bf lo = *(const v8bf*)p;
  v8bf hi = *(const v8bf*)(p + 16);
  return __builtin_shufflevector(lo, hi, 0,1,2,3,4,5,6,7,8,9,10,11,12,13,14,15);
}
// B: 32x16 (KxN). lane<16: N=lane, K = k0 + {0..7, 16..23}; lane>=16: +8.
__device__ __forceinline__ v16bf frag_b(const bf16_t* buf, int ldk, int n_base,
                                        int k0, int lane) {
  const int half = (lane >> 4) & 1;
  const int n = n_base + (lane & 15);
  const bf16_t* p = buf + n * ldk + k0 + half * 8;
  v8bf lo = *(const v8bf*)p;
  v8bf hi = *(const v8bf*)(p + 16);
  return __builtin_shufflevector(lo, hi, 0,1,2,3,4,5,6,7,8,9,10,11,12,13,14,15);
}

__device__ __forceinline__ v8f wmma_bf16(v16bf a, v16bf b, v8f c) {
  return __builtin_amdgcn_wmma_f32_16x16x32_bf16(false, a, false, b,
                                                 (short)0, c, false, false);
}

// low 32 bits of a generic pointer to LDS == LDS byte offset (flat aperture)
__device__ __forceinline__ unsigned lds_off(const void* p) {
  return (unsigned)(unsigned long long)p;
}

// ---- Tensor Data Mover: async 2D tile (64 x 128 elems, 2B) global -> LDS --
// D# per CDNA5 ISA ch.8: group0 {count=1, lds_addr, global_addr[56:0], type=2}
//                       group1 {data_size=1(2B), tensor_dim0=2048, stride0=2048,
//                               tensor_dim1=rows, tile_dim0=64, tile_dim1=128}
// 2D tensor -> 2-SGPR-group form (VADDR2/3 = NULL). Tracked by TENSORcnt.
__device__ __forceinline__ void tdm_load_tile(unsigned lds_addr,
                                              unsigned long long gaddr,
                                              int rows) {
  si4 g0;
  g0[0] = 1;                                            // count=1, user D#
  g0[1] = (int)lds_addr;                                // lds_addr (bytes)
  g0[2] = (int)(unsigned)gaddr;                         // global_addr[31:0]
  g0[3] = (int)(((unsigned)(gaddr >> 32) & 0x01FFFFFFu) // global_addr[56:32]
                | 0x80000000u);                         // type=2 ("image")
  si8 g1;
  g1[0] = 0x00010000;                                   // data_size=1 -> 2B
  g1[1] = (int)(0x0800u << 16);                         // tensor_dim0 = 2048
  g1[2] = (int)(((unsigned)rows & 0xFFFFu) << 16);      // tensor_dim1 lo16
  g1[3] = (int)((((unsigned)rows >> 16) & 0xFFFFu)      // tensor_dim1 hi16
                | (64u << 16));                         // tile_dim0 = 64
  g1[4] = 128;                                          // tile_dim1 = 128
  g1[5] = 2048;                                         // tensor_dim0_stride
  g1[6] = 0;
  g1[7] = 0;
  asm volatile("tensor_load_to_lds %0, %1" :: "s"(g0), "s"(g1) : "memory");
}

// ---------------------------------------------------------------------------
// prep: fp32 -> bf16 weight conversion (once per launch), zero ctx0
// ---------------------------------------------------------------------------
__global__ __launch_bounds__(256) void prep_kernel(
    const float* __restrict__ wcls, const float* __restrict__ wih,
    const float* __restrict__ whh,
    bf16_t* __restrict__ wcls_b, bf16_t* __restrict__ wih_b,
    bf16_t* __restrict__ whh_b, float* __restrict__ ctx) {
  const size_t tid = (size_t)blockIdx.x * 256u + threadIdx.x;
  const size_t stride = (size_t)gridDim.x * 256u;
  for (size_t i = tid; i < (size_t)DV * KCLS; i += stride) wcls_b[i] = (bf16_t)wcls[i];
  for (size_t i = tid; i < (size_t)D3H * KIH; i += stride)  wih_b[i]  = (bf16_t)wih[i];
  for (size_t i = tid; i < (size_t)D3H * DH;  i += stride)  whh_b[i]  = (bf16_t)whh[i];
  for (size_t i = tid; i < (size_t)DB * DH;   i += stride)  ctx[i]    = 0.f;
}

// ---------------------------------------------------------------------------
// h0 = initial_hidden_state @ W_hm^T + b_hm   (one-time, 33 MFLOP)
// ---------------------------------------------------------------------------
__global__ __launch_bounds__(256) void init_h0_kernel(
    const float* __restrict__ ihs, const float* __restrict__ whm,
    const float* __restrict__ bhm, float* __restrict__ h) {
  const int idx = blockIdx.x * 256 + threadIdx.x;   // 32*1024 total
  const int b = idx >> 10, n = idx & 1023;
  const float* x = ihs + b * DH;
  const float* w = whm + (size_t)n * DH;
  float acc = bhm[n];
  for (int k = 0; k < DH; ++k) acc = fmaf(x[k], w[k], acc);
  h[idx] = acc;
}

// ---------------------------------------------------------------------------
// Per-step gate GEMMs (WMMA): grid.y==0 -> gi = rnn_in @ W_ih^T   (K=1536)
//                             grid.y==1 -> gh = h      @ W_hh^T   (K=1024)
// rnn_in is gathered inline: [emb[tgt[b,t]] (512) | ctx (1024)] -> bf16 LDS
// ---------------------------------------------------------------------------
#define G_NT 128
#define G_KC 64
__global__ __launch_bounds__(128) void gru_gates_kernel(
    const float* __restrict__ emb, const int* __restrict__ tgt, int t,
    const float* __restrict__ ctx, const float* __restrict__ h,
    const bf16_t* __restrict__ wih_b, const bf16_t* __restrict__ whh_b,
    float* __restrict__ gi, float* __restrict__ gh) {
  __shared__ __align__(16) bf16_t As[DB][G_KC];
  __shared__ __align__(16) bf16_t Bs[G_NT][G_KC];
  const int which = blockIdx.y;
  const int K = which ? DH : KIH;
  const bf16_t* W = which ? whh_b : wih_b;
  float* outp = which ? gh : gi;
  const int tid = threadIdx.x, wid = tid >> 5, lane = tid & 31;
  const int n_glob = blockIdx.x * G_NT;
  v8f acc[2][2];
  #pragma unroll
  for (int i = 0; i < 2; ++i)
    #pragma unroll
    for (int j = 0; j < 2; ++j) acc[i][j] = (v8f){0,0,0,0,0,0,0,0};

  for (int kc = 0; kc < K; kc += G_KC) {
    for (int i = tid; i < DB * G_KC; i += 128) {
      const int b = i / G_KC, k = kc + (i % G_KC);
      float v;
      if (which) {
        v = h[b * DH + k];
      } else if (k < DE) {
        v = emb[(size_t)tgt[b * DT + t] * DE + k];
      } else {
        v = ctx[b * DH + (k - DE)];
      }
      As[b][i % G_KC] = (bf16_t)v;
    }
    for (int i = tid; i < G_NT * (G_KC / 8); i += 128) {
      const int r = i >> 3, c = (i & 7) * 8;
      *(v8bf*)&Bs[r][c] = *(const v8bf*)&W[(size_t)(n_glob + r) * K + kc + c];
    }
    __syncthreads();
    #pragma unroll
    for (int ks = 0; ks < G_KC; ks += 32) {
      v16bf a0 = frag_a(&As[0][0], G_KC, 0, ks, lane);
      v16bf a1 = frag_a(&As[0][0], G_KC, 16, ks, lane);
      #pragma unroll
      for (int j = 0; j < 2; ++j) {
        v16bf bfr = frag_b(&Bs[0][0], G_KC, wid * 32 + j * 16, ks, lane);
        acc[0][j] = wmma_bf16(a0, bfr, acc[0][j]);
        acc[1][j] = wmma_bf16(a1, bfr, acc[1][j]);
      }
    }
    __syncthreads();
  }
  const int half = lane >> 4, lm = lane & 15;
  #pragma unroll
  for (int mi = 0; mi < 2; ++mi)
    #pragma unroll
    for (int j = 0; j < 2; ++j) {
      const int n = n_glob + wid * 32 + j * 16 + lm;
      #pragma unroll
      for (int i = 0; i < 8; ++i) {
        const int m = mi * 16 + half * 8 + i;
        outp[m * D3H + n] = acc[mi][j][i];
      }
    }
}

// ---------------------------------------------------------------------------
// Per-step fused gates + dot-attention. One workgroup per batch row.
// ---------------------------------------------------------------------------
__global__ __launch_bounds__(128) void gru_attn_kernel(
    const float* __restrict__ gi, const float* __restrict__ gh,
    const float* __restrict__ bih, const float* __restrict__ bhh,
    const float* __restrict__ enc,  // [B][S][H]
    float* __restrict__ h, float* __restrict__ ctx,
    bf16_t* __restrict__ pred_t) {  // [B][2H] slice for this timestep
  __shared__ float hs[DH];
  __shared__ float sc[DS];
  __shared__ float rbuf[DS];
  const int b = blockIdx.x, tid = threadIdx.x;

  for (int j = tid; j < DH; j += 128) {
    const float ir = gi[b * D3H + j]            + bih[j];
    const float iz = gi[b * D3H + DH + j]       + bih[DH + j];
    const float in = gi[b * D3H + 2 * DH + j]   + bih[2 * DH + j];
    const float hr = gh[b * D3H + j]            + bhh[j];
    const float hz = gh[b * D3H + DH + j]       + bhh[DH + j];
    const float hn = gh[b * D3H + 2 * DH + j]   + bhh[2 * DH + j];
    const float r = 1.f / (1.f + __expf(-(ir + hr)));
    const float z = 1.f / (1.f + __expf(-(iz + hz)));
    const float n = tanhf(in + r * hn);
    hs[j] = (1.f - z) * n + z * h[b * DH + j];
  }
  __syncthreads();

  {
    const float* e = enc + ((size_t)b * DS + tid) * DH;
    float a = 0.f;
    for (int j = 0; j < DH; ++j) a = fmaf(e[j], hs[j], a);
    sc[tid] = a;
  }
  __syncthreads();
  rbuf[tid] = sc[tid];
  __syncthreads();
  for (int off = 64; off > 0; off >>= 1) {
    if (tid < off) rbuf[tid] = fmaxf(rbuf[tid], rbuf[tid + off]);
    __syncthreads();
  }
  const float mx = rbuf[0];
  __syncthreads();
  const float ex = __expf(sc[tid] - mx);
  rbuf[tid] = ex;
  __syncthreads();
  for (int off = 64; off > 0; off >>= 1) {
    if (tid < off) rbuf[tid] += rbuf[tid + off];
    __syncthreads();
  }
  const float inv = 1.f / rbuf[0];
  __syncthreads();
  sc[tid] = ex * inv;
  __syncthreads();

  for (int j = tid; j < DH; j += 128) {
    float a = 0.f;
    for (int s = 0; s < DS; ++s)
      a = fmaf(sc[s], enc[((size_t)b * DS + s) * DH + j], a);
    const float hn = hs[j];
    ctx[b * DH + j] = a;
    h[b * DH + j] = hn;
    pred_t[b * KCLS + j] = (bf16_t)a;
    pred_t[b * KCLS + DH + j] = (bf16_t)hn;
  }
}

// ---------------------------------------------------------------------------
// Batched classifier: [T*B=2048, 2048](bf16) x W_cls^T(bf16, L2-resident)
//   -> d_out[b][t][v] (fp32). 128x128 tiles, 8 waves, K-chunk 64.
// TDM double-buffered: wave0 issues tensor_load_to_lds for chunk k+1 while
// all waves run WMMA on chunk k; s_wait_tensorcnt + barrier closes the stage.
// ---------------------------------------------------------------------------
#define C_MT 128
#define C_NT 128
#define C_KC 64
__global__ __launch_bounds__(256) void cls_gemm_kernel(
    const bf16_t* __restrict__ pred,   // [2048][2048], row = t*B + b
    const bf16_t* __restrict__ wcls_b, // [32000][2048]
    const float* __restrict__ bcls, float* __restrict__ out) {
  __shared__ __align__(128) bf16_t As[2][C_MT][C_KC];  // 2 x 16 KB
  __shared__ __align__(128) bf16_t Bs[2][C_NT][C_KC];  // 2 x 16 KB
  const int nblk = blockIdx.x, mblk = blockIdx.y;
  const int tid = threadIdx.x, wid = tid >> 5, lane = tid & 31;
  const int wm = wid >> 1;  // 4 wave-rows x 32 rows
  const int wn = wid & 1;   // 2 wave-cols x 64 cols
  const int m_glob = mblk * C_MT, n_glob = nblk * C_NT;

  const unsigned long long a_base =
      (unsigned long long)pred + (size_t)m_glob * KCLS * 2;
  const unsigned long long b_base =
      (unsigned long long)wcls_b + (size_t)n_glob * KCLS * 2;
  const unsigned a_lds[2] = {lds_off(&As[0][0][0]), lds_off(&As[1][0][0])};
  const unsigned b_lds[2] = {lds_off(&Bs[0][0][0]), lds_off(&Bs[1][0][0])};

  v8f acc[2][4];
  #pragma unroll
  for (int i = 0; i < 2; ++i)
    #pragma unroll
    for (int j = 0; j < 4; ++j) acc[i][j] = (v8f){0,0,0,0,0,0,0,0};

  // prologue: chunk 0 into buffer 0
  if (wid == 0) {
    tdm_load_tile(a_lds[0], a_base, 2048);
    tdm_load_tile(b_lds[0], b_base, DV);
    __builtin_amdgcn_s_wait_tensorcnt(0);
  }
  __syncthreads();

  int p = 0;
  for (int kc = 0; kc < KCLS; kc += C_KC) {
    if (wid == 0 && (kc + C_KC) < KCLS) {  // kick next chunk into other buffer
      const unsigned long long koff = (unsigned long long)(kc + C_KC) * 2;
      tdm_load_tile(a_lds[p ^ 1], a_base + koff, 2048);
      tdm_load_tile(b_lds[p ^ 1], b_base + koff, DV);
    }
    const bf16_t* Ap = &As[p][0][0];
    const bf16_t* Bp = &Bs[p][0][0];
    #pragma unroll
    for (int ks = 0; ks < C_KC; ks += 32) {
      v16bf a0 = frag_a(Ap, C_KC, wm * 32, ks, lane);
      v16bf a1 = frag_a(Ap, C_KC, wm * 32 + 16, ks, lane);
      #pragma unroll
      for (int j = 0; j < 4; ++j) {
        v16bf bfr = frag_b(Bp, C_KC, wn * 64 + j * 16, ks, lane);
        acc[0][j] = wmma_bf16(a0, bfr, acc[0][j]);
        acc[1][j] = wmma_bf16(a1, bfr, acc[1][j]);
      }
    }
    if (wid == 0) __builtin_amdgcn_s_wait_tensorcnt(0);
    __syncthreads();
    p ^= 1;
  }

  // epilogue: remap row m = t*B + b -> out[b][t][v], add bias
  const int half = lane >> 4, lm = lane & 15;
  #pragma unroll
  for (int mi = 0; mi < 2; ++mi)
    #pragma unroll
    for (int j = 0; j < 4; ++j) {
      const int n = n_glob + wn * 64 + j * 16 + lm;
      const float bias = bcls[n];
      #pragma unroll
      for (int i = 0; i < 8; ++i) {
        const int m = m_glob + wm * 32 + mi * 16 + half * 8 + i;
        const int t = m >> 5, b = m & 31;  // row = t*32 + b
        out[((size_t)b * DT + t) * DV + n] = acc[mi][j][i] + bias;
      }
    }
}

// ---------------------------------------------------------------------------
// host-side orchestration (graph-capture safe: only stream launches)
// ---------------------------------------------------------------------------
extern "C" void kernel_launch(void* const* d_in, const int* in_sizes, int n_in,
                              void* d_out, int out_size, void* d_ws, size_t ws_size,
                              hipStream_t stream) {
  (void)in_sizes; (void)n_in; (void)out_size; (void)ws_size;
  const float* enc  = (const float*)d_in[0];   // [B,S,H]
  const float* ihs  = (const float*)d_in[1];   // [B,H]
  const int*   tgt  = (const int*)  d_in[2];   // [B,T]
  const float* emb  = (const float*)d_in[3];   // [V,E]
  const float* wih  = (const float*)d_in[4];   // [3H, E+H]
  const float* whh  = (const float*)d_in[5];   // [3H, H]
  const float* bih  = (const float*)d_in[6];
  const float* bhh  = (const float*)d_in[7];
  const float* whm  = (const float*)d_in[8];   // [H,H]
  const float* bhm  = (const float*)d_in[9];
  const float* wcls = (const float*)d_in[10];  // [V, 2H]
  const float* bcls = (const float*)d_in[11];
  float* out = (float*)d_out;

  // workspace carve-up
  char* ws = (char*)d_ws;
  size_t off = 0;
  bf16_t* wcls_b = (bf16_t*)(ws + off); off += (size_t)DV * KCLS * 2;      // 131.07 MB
  bf16_t* wih_b  = (bf16_t*)(ws + off); off += (size_t)D3H * KIH * 2;      //   9.44 MB
  bf16_t* whh_b  = (bf16_t*)(ws + off); off += (size_t)D3H * DH * 2;       //   6.29 MB
  bf16_t* pred   = (bf16_t*)(ws + off); off += (size_t)DT * DB * KCLS * 2; //   8.39 MB
  float*  hbuf   = (float*) (ws + off); off += (size_t)DB * DH * 4;
  float*  ctxbuf = (float*) (ws + off); off += (size_t)DB * DH * 4;
  float*  gibuf  = (float*) (ws + off); off += (size_t)DB * D3H * 4;
  float*  ghbuf  = (float*) (ws + off); off += (size_t)DB * D3H * 4;

  prep_kernel<<<4096, 256, 0, stream>>>(wcls, wih, whh, wcls_b, wih_b, whh_b, ctxbuf);
  init_h0_kernel<<<(DB * DH) / 256, 256, 0, stream>>>(ihs, whm, bhm, hbuf);

  for (int t = 0; t < DT; ++t) {
    gru_gates_kernel<<<dim3(D3H / G_NT, 2), 128, 0, stream>>>(
        emb, tgt, t, ctxbuf, hbuf, wih_b, whh_b, gibuf, ghbuf);
    gru_attn_kernel<<<DB, 128, 0, stream>>>(
        gibuf, ghbuf, bih, bhh, enc, hbuf, ctxbuf, pred + (size_t)t * DB * KCLS);
  }

  cls_gemm_kernel<<<dim3(DV / C_NT, (DT * DB) / C_MT), 256, 0, stream>>>(
      pred, wcls_b, bcls, out);
}
